// SuperPoint_9929964388436
// MI455X (gfx1250) — matrix operand order repaired and checked by reading, not compiled, not run on hardware
//
#include <hip/hip_runtime.h>
#include <hip/hip_bf16.h>
#include <math.h>
#include <stdint.h>

// ---------------------------------------------------------------------------
// SuperPoint forward on gfx1250: implicit-GEMM convs via v_wmma_f32_16x16x32_f16
// - Register-blocked: each wave computes 16x32 output via 2 accumulators,
//   K-chunks of 64 staged in LDS -> 4 WMMA issues per staging pass.
// - Weight tile staged with gfx1250 async global->LDS DMA
//   (global_load_async_to_lds_b128 + s_wait_asynccnt).
// ---------------------------------------------------------------------------

typedef __attribute__((ext_vector_type(16))) _Float16 v16h;
typedef __attribute__((ext_vector_type(8)))  float    v8f;

#define BATCH 4
#define H0 480
#define W0 640
#define HC 60
#define WC 80
#define MAX_KP 1024
#define CAND_CAP 16384
#define KP_THR 0.005f
#define NMS_R 4
#define BORDER 4

#define TM 64   // Cout tile per block
#define TN 64   // pixels per block (one output row segment)
#define TKC 64  // K chunk staged per pass (2 WMMA-depth sub-chunks)

// -------- weight conversion: fp32 [Cout][K] -> f16 [Mpad][Kpad], zero-padded -
__global__ void k_convert_weights(const float* __restrict__ w,
                                  _Float16* __restrict__ w16,
                                  int Cout, int K, int Kpad, int Mpad) {
  int i = blockIdx.x * blockDim.x + threadIdx.x;
  int total = Mpad * Kpad;
  if (i >= total) return;
  int m = i / Kpad, k = i % Kpad;
  w16[i] = (m < Cout && k < K) ? (_Float16)w[(size_t)m * K + k] : (_Float16)0.0f;
}

// ---------------- fp32 -> f16 elementwise ------------------------------------
__global__ void k_f32_to_f16(const float* __restrict__ in,
                             _Float16* __restrict__ out, int n) {
  int i = blockIdx.x * blockDim.x + threadIdx.x;
  if (i < n) out[i] = (_Float16)in[i];
}

// ---------------- implicit-GEMM conv with WMMA -------------------------------
// out[b][m][oy][ox] = relu( sum_k A[m][k] * im2col[k][pix] + bias[m] )
// K = Cin*ks*ks ordered (ci, ky, kx). Weights padded to Mpad x Kpad.
// Block tile: M=64 x N=64. 8 waves = 4 (M) x 2 (N); each wave: 1 M-subtile x
// 2 N-subtiles -> 2 accumulators, A-fragment reused across both.
__global__ __launch_bounds__(256) void k_conv_wmma(
    const _Float16* __restrict__ in,    // [B][Cin][H][W]
    const _Float16* __restrict__ wgt,   // [Mpad][Kpad] zero-padded
    const float*    __restrict__ bias,  // [Cout]
    _Float16* __restrict__ out16,       // NCHW f16 (if !out_f32)
    float*    __restrict__ outf,        // NCHW f32 (if out_f32)
    int Cin, int H, int W, int Cout,
    int ks, int pad, int relu, int out_f32, int tilesX) {
  __shared__ __align__(16) _Float16 Asm[TM][TKC];   // 8 KB
  __shared__ __align__(16) _Float16 Bsm[TKC][TN];   // 8 KB

  const int tid  = threadIdx.x;
  const int lane = tid & 31;
  const int wave = tid >> 5;
  const int lr = lane & 15;
  const int hi = lane >> 4;

  const int K    = Cin * ks * ks;
  const int Kpad = (K + TKC - 1) / TKC * TKC;
  const int tx = blockIdx.x % tilesX;
  const int oy = blockIdx.x / tilesX;
  const int mBase  = blockIdx.y * TM;
  const int b      = blockIdx.z;
  const int oxBase = tx * TN;

  const int mSub  = (wave >> 1) * 16;  // 4 wave-rows in M
  const int nBase = (wave & 1)  * 32;  // 2 wave-cols, 32 px each (2 subtiles)

  v8f acc0 = {0.f,0.f,0.f,0.f,0.f,0.f,0.f,0.f};
  v8f acc1 = {0.f,0.f,0.f,0.f,0.f,0.f,0.f,0.f};

  const size_t inPlane = (size_t)H * W;
  const int ksq = ks * ks;

  // LDS byte offset of Asm: low 32 bits of the generic address of a __shared__
  // object are the wave's LDS offset on AMDGPU.
  const unsigned ldsA0 = (unsigned)(uintptr_t)(&Asm[0][0]);

  for (int k0 = 0; k0 < Kpad; k0 += TKC) {
    // prefetch next weight chunk (gfx1250 global_prefetch_b8 path)
    if (k0 + TKC < Kpad)
      __builtin_prefetch(&wgt[(size_t)(mBase + mSub) * Kpad + k0 + TKC], 0, 3);

    // ---- stage A tile (64 x 64 halves = 8 KB) via async DMA to LDS ----------
    // 512 x b128 transfers, 2 per thread. Rows are Kpad-strided (Kpad % 64 == 0
    // so every 16-byte transfer is aligned); matrix is zero-padded so no guards.
    for (int seg = tid; seg < (TM * TKC) / 8; seg += 256) {
      int m = seg >> 3;            // 8 segments of 8 halves per 64-half row
      int k = (seg & 7) << 3;
      unsigned lofs = ldsA0 + (unsigned)((m * TKC + k) * 2);
      const _Float16* gp = wgt + (size_t)(mBase + m) * Kpad + (k0 + k);
      asm volatile("global_load_async_to_lds_b128 %0, %1, off"
                   :: "v"(lofs), "v"((unsigned long long)(uintptr_t)gp)
                   : "memory");
    }

    // ---- stage B tile (im2col, 64 x 64 halves) with boundary-zero gather ----
    for (int i = tid; i < TKC * TN; i += 256) {
      int k = i / TN, n = i % TN;
      int gk = k0 + k;
      _Float16 v = (_Float16)0.0f;
      if (gk < K) {
        int ci  = gk / ksq;
        int rem = gk % ksq;
        int ky = rem / ks, kx = rem % ks;
        int iy = oy + ky - pad;
        int ix = oxBase + n + kx - pad;
        if (iy >= 0 && iy < H && ix >= 0 && ix < W)
          v = in[((size_t)b * Cin + ci) * inPlane + (size_t)iy * W + ix];
      }
      Bsm[k][n] = v;
    }

    // wait for this wave's async DMA, then block barrier for cross-wave LDS
    asm volatile("s_wait_asynccnt 0x0" ::: "memory");
    __syncthreads();

    // Two 32-deep sub-chunks; A fragment reused across both N subtiles.
#pragma unroll
    for (int s = 0; s < 2; ++s) {
      const int kOff = s * 32;
      // A fragment: lane lr holds row M=lr; elem e -> k = (e>>3)*16+hi*8+(e&7)
      v16h af, bf0, bf1;
#pragma unroll
      for (int e = 0; e < 16; ++e) {
        int kl = kOff + ((e >> 3) << 4) + (hi << 3) + (e & 7);
        af[e]  = Asm[mSub + lr][kl];
        // B fragment: lane holds column N = lr; elem e -> same k as A
        bf0[e] = Bsm[kl][nBase + lr];
        bf1[e] = Bsm[kl][nBase + 16 + lr];
      }
      acc0 = __builtin_amdgcn_wmma_f32_16x16x32_f16(
          false, af, false, bf0, (short)0, acc0, false, false);
      acc1 = __builtin_amdgcn_wmma_f32_16x16x32_f16(
          false, af, false, bf1, (short)0, acc1, false, false);
    }
    __syncthreads();
  }

  // C/D layout: VGPR r -> M = r + 8*hi, N = lr
  const size_t outPlane = (size_t)H * W;
  const int m0 = mBase + mSub + (hi << 3);
#pragma unroll
  for (int r = 0; r < 8; ++r) {
    int m = m0 + r;
    if (m >= Cout || oy >= H) continue;
    float bv = bias[m];
    size_t rowOff = ((size_t)b * Cout + m) * outPlane + (size_t)oy * W;
    int n0 = oxBase + nBase + lr;
    int n1 = n0 + 16;
    if (n0 < W) {
      float v = acc0[r] + bv;
      if (relu) v = fmaxf(v, 0.0f);
      if (out_f32) outf[rowOff + n0] = v;
      else         out16[rowOff + n0] = (_Float16)v;
    }
    if (n1 < W) {
      float v = acc1[r] + bv;
      if (relu) v = fmaxf(v, 0.0f);
      if (out_f32) outf[rowOff + n1] = v;
      else         out16[rowOff + n1] = (_Float16)v;
    }
  }
}

// ---------------- 2x2 max-pool stride 2 on f16 NCHW --------------------------
__global__ void k_pool2(const _Float16* __restrict__ in, _Float16* __restrict__ out,
                        int BC, int H, int W) {
  int Ho = H >> 1, Wo = W >> 1;
  int i = blockIdx.x * blockDim.x + threadIdx.x;
  if (i >= BC * Ho * Wo) return;
  int bc = i / (Ho * Wo);
  int p  = i % (Ho * Wo);
  int oy = p / Wo, ox = p % Wo;
  const _Float16* s = in + (size_t)bc * H * W;
  float a = (float)s[(size_t)(2 * oy) * W + 2 * ox];
  float b = (float)s[(size_t)(2 * oy) * W + 2 * ox + 1];
  float c = (float)s[(size_t)(2 * oy + 1) * W + 2 * ox];
  float d = (float)s[(size_t)(2 * oy + 1) * W + 2 * ox + 1];
  float m = fmaxf(fmaxf(a, b), fmaxf(c, d));
  out[(size_t)bc * Ho * Wo + p] = (_Float16)m;
}

// ---------------- softmax over 65 ch + drop dustbin + pixel_shuffle(8) -------
__global__ void k_softmax_shuffle(const float* __restrict__ logits,
                                  float* __restrict__ smap,
                                  int h, int w) {
  int i = blockIdx.x * blockDim.x + threadIdx.x;
  int total = BATCH * h * w;
  if (i >= total) return;
  int b = i / (h * w);
  int p = i % (h * w);
  int y = p / w, x = p % w;
  const float* L = logits + (size_t)b * 65 * h * w + p;
  size_t cs = (size_t)h * w;
  float mx = -1e30f;
  for (int c = 0; c < 65; ++c) mx = fmaxf(mx, L[c * cs]);
  float sum = 0.f;
  for (int c = 0; c < 65; ++c) sum += __expf(L[c * cs] - mx);
  float inv = 1.f / sum;
  int H8 = h * 8, W8 = w * 8;
  for (int c = 0; c < 64; ++c) {
    int r = c >> 3, cc = c & 7;
    smap[((size_t)b * H8 + (y * 8 + r)) * W8 + (x * 8 + cc)] =
        __expf(L[c * cs] - mx) * inv;
  }
}

// ---------------- separable (2r+1) max-pool, -inf padding --------------------
__global__ void k_mp_row(const float* __restrict__ in, float* __restrict__ out,
                         int BH, int W, int r) {
  int i = blockIdx.x * blockDim.x + threadIdx.x;
  if (i >= BH * W) return;
  int x = i % W;
  const float* row = in + (i - x);
  float m = -1e30f;
  for (int d = -r; d <= r; ++d) {
    int xx = x + d;
    if (xx >= 0 && xx < W) m = fmaxf(m, row[xx]);
  }
  out[i] = m;
}
__global__ void k_mp_col(const float* __restrict__ in, float* __restrict__ out,
                         int B, int H, int W, int r) {
  int i = blockIdx.x * blockDim.x + threadIdx.x;
  if (i >= B * H * W) return;
  int b = i / (H * W);
  int p = i % (H * W);
  int y = p / W, x = p % W;
  const float* img = in + (size_t)b * H * W;
  float m = -1e30f;
  for (int d = -r; d <= r; ++d) {
    int yy = y + d;
    if (yy >= 0 && yy < H) m = fmaxf(m, img[(size_t)yy * W + x]);
  }
  out[i] = m;
}

// ---------------- NMS elementwise steps --------------------------------------
__global__ void k_eqmask(const float* __restrict__ s, const float* __restrict__ mp,
                         float* __restrict__ mask, int n) {
  int i = blockIdx.x * blockDim.x + threadIdx.x;
  if (i < n) mask[i] = (fabsf(s[i] - mp[i]) < 1e-6f) ? 1.f : 0.f;
}
__global__ void k_supp(const float* __restrict__ mpmask, const float* __restrict__ s,
                       float* __restrict__ supp, float* __restrict__ ss, int n) {
  int i = blockIdx.x * blockDim.x + threadIdx.x;
  if (i >= n) return;
  float sp = (mpmask[i] > 0.f) ? 1.f : 0.f;
  supp[i] = sp;
  ss[i] = (sp > 0.f) ? 0.f : s[i];
}
__global__ void k_update(const float* __restrict__ ss, const float* __restrict__ mpss,
                         const float* __restrict__ supp, float* __restrict__ mask, int n) {
  int i = blockIdx.x * blockDim.x + threadIdx.x;
  if (i >= n) return;
  float nm = (fabsf(ss[i] - mpss[i]) < 1e-6f) ? 1.f : 0.f;
  mask[i] = (mask[i] > 0.f || (nm > 0.f && supp[i] == 0.f)) ? 1.f : 0.f;
}
__global__ void k_final(const float* __restrict__ s, const float* __restrict__ mask,
                        float* __restrict__ out, int n) {
  int i = blockIdx.x * blockDim.x + threadIdx.x;
  if (i < n) out[i] = (mask[i] > 0.f) ? s[i] : 0.f;
}

// ---------------- candidate compaction ---------------------------------------
__global__ void k_zero_ints(int* p, int n) {
  int i = blockIdx.x * blockDim.x + threadIdx.x;
  if (i < n) p[i] = 0;
}
__global__ void k_compact(const float* __restrict__ nms,
                          float* __restrict__ cval, int* __restrict__ cidx,
                          int* __restrict__ cnt, int H, int W) {
  int i = blockIdx.x * blockDim.x + threadIdx.x;
  if (i >= BATCH * H * W) return;
  int b = i / (H * W);
  int p = i % (H * W);
  int y = p / W, x = p % W;
  float s = nms[i];
  if (s > KP_THR && y >= BORDER && y < H - BORDER && x >= BORDER && x < W - BORDER) {
    int j = atomicAdd(&cnt[b], 1);
    if (j < CAND_CAP) {
      cval[(size_t)b * CAND_CAP + j] = s;
      cidx[(size_t)b * CAND_CAP + j] = p;
    }
  }
}

// ---------------- iterative top-K selection (one block per image) ------------
__global__ __launch_bounds__(256) void k_select_topk(
    float* __restrict__ cval, const int* __restrict__ cidx, const int* __restrict__ cnt,
    float* __restrict__ kp, float* __restrict__ scores, float* __restrict__ valid,
    int W) {
  int b = blockIdx.x;
  int n = cnt[b];
  if (n > CAND_CAP) n = CAND_CAP;
  float* v = cval + (size_t)b * CAND_CAP;
  const int* ix = cidx + (size_t)b * CAND_CAP;
  __shared__ float sv[256];
  __shared__ int si[256];
  for (int it = 0; it < MAX_KP; ++it) {
    float best = -1e30f; int bi = -1;
    for (int j = threadIdx.x; j < n; j += 256) {
      float x = v[j];
      if (x > best || (x == best && bi >= 0 && ix[j] < ix[bi])) { best = x; bi = j; }
    }
    sv[threadIdx.x] = best; si[threadIdx.x] = bi;
    __syncthreads();
    for (int s2 = 128; s2 > 0; s2 >>= 1) {
      if (threadIdx.x < s2) {
        float ov = sv[threadIdx.x + s2]; int oi = si[threadIdx.x + s2];
        float mv = sv[threadIdx.x];      int mi = si[threadIdx.x];
        bool take = (oi >= 0) && (mi < 0 || ov > mv ||
                                  (ov == mv && ix[oi] < ix[mi]));
        if (take) { sv[threadIdx.x] = ov; si[threadIdx.x] = oi; }
      }
      __syncthreads();
    }
    if (threadIdx.x == 0) {
      float val = sv[0]; int j = si[0];
      float kx = 0.f, ky = 0.f, sc = 0.f, vd = 0.f;
      if (j >= 0) {
        int p = ix[j];
        kx = (float)(p % W);
        ky = (float)(p / W);
        v[j] = -2e30f;
        if (val > KP_THR) { sc = val; vd = 1.f; }
      }
      kp[((size_t)b * MAX_KP + it) * 2 + 0] = kx;
      kp[((size_t)b * MAX_KP + it) * 2 + 1] = ky;
      scores[(size_t)b * MAX_KP + it] = sc;
      valid [(size_t)b * MAX_KP + it] = vd;
    }
    __syncthreads();
  }
}

// ---------------- per-pixel L2 normalize of descriptor map (in place) --------
__global__ void k_norm_map(float* __restrict__ d, int C, int HW) {
  int i = blockIdx.x * blockDim.x + threadIdx.x;
  if (i >= BATCH * HW) return;
  int b = i / HW, p = i % HW;
  float ss = 0.f;
  for (int c = 0; c < C; ++c) {
    float x = d[((size_t)b * C + c) * HW + p];
    ss += x * x;
  }
  float inv = 1.f / fmaxf(sqrtf(ss), 1e-12f);
  for (int c = 0; c < C; ++c)
    d[((size_t)b * C + c) * HW + p] *= inv;
}

// ---------------- bilinear descriptor sampling + renormalize -----------------
__global__ __launch_bounds__(256) void k_sample_desc(
    const float* __restrict__ descn,    // [B][256][HC][WC], pre-normalized
    const float* __restrict__ kp,       // [B][MAX_KP][2]
    float* __restrict__ dout) {         // [B][256][MAX_KP]
  int b = blockIdx.x / MAX_KP;
  int k = blockIdx.x % MAX_KP;
  int c = threadIdx.x;     // 256 channels
  float kx = kp[((size_t)b * MAX_KP + k) * 2 + 0];
  float ky = kp[((size_t)b * MAX_KP + k) * 2 + 1];
  const float s = 8.f;
  float nx = ((kx - s * 0.5f + 0.5f) / (WC * s - s * 0.5f - 0.5f)) * 2.f - 1.f;
  float ny = ((ky - s * 0.5f + 0.5f) / (HC * s - s * 0.5f - 0.5f)) * 2.f - 1.f;
  float x = (nx + 1.f) * 0.5f * (WC - 1);
  float y = (ny + 1.f) * 0.5f * (HC - 1);
  float x0f = floorf(x), y0f = floorf(y);
  float wa = (x0f + 1.f - x) * (y0f + 1.f - y);
  float wb = (x0f + 1.f - x) * (y - y0f);
  float wc = (x - x0f) * (y0f + 1.f - y);
  float wd = (x - x0f) * (y - y0f);
  int x0 = (int)x0f, y0 = (int)y0f;
  const float* ch = descn + ((size_t)b * 256 + c) * (HC * WC);
  auto fetch = [&](int yy, int xx) -> float {
    if (yy < 0 || yy >= HC || xx < 0 || xx >= WC) return 0.f;
    return ch[(size_t)yy * WC + xx];
  };
  float o = fetch(y0, x0) * wa + fetch(y0 + 1, x0) * wb +
            fetch(y0, x0 + 1) * wc + fetch(y0 + 1, x0 + 1) * wd;
  __shared__ float red[256];
  red[c] = o * o;
  __syncthreads();
  for (int s2 = 128; s2 > 0; s2 >>= 1) {
    if (c < s2) red[c] += red[c + s2];
    __syncthreads();
  }
  float nrm = fmaxf(sqrtf(red[0]), 1e-12f);
  dout[((size_t)b * 256 + c) * MAX_KP + k] = o / nrm;
}

// ===========================================================================
// Host orchestration
// ===========================================================================
static inline size_t al256(size_t x) { return (x + 255) & ~(size_t)255; }

extern "C" void kernel_launch(void* const* d_in, const int* in_sizes, int n_in,
                              void* d_out, int out_size, void* d_ws, size_t ws_size,
                              hipStream_t stream) {
  (void)in_sizes; (void)n_in; (void)out_size; (void)ws_size;
  const float* image = (const float*)d_in[0];
  // weights/biases in setup order
  const float* W_[12]; const float* Bb[12];
  for (int j = 0; j < 12; ++j) { W_[j] = (const float*)d_in[1 + 2 * j]; Bb[j] = (const float*)d_in[2 + 2 * j]; }
  // layer specs: {Cout, Cin, ks}
  const int CO[12] = {64,64,64,64,128,128,128,128,256,65,256,256};
  const int CI[12] = {1,64,64,64,64,128,128,128,128,256,128,256};
  const int KS[12] = {3,3,3,3,3,3,3,3,3,1,3,1};

  // ---- workspace bump allocator -------------------------------------------
  char* ws = (char*)d_ws;
  size_t off = 0;
  auto alloc = [&](size_t bytes) -> void* { void* p = ws + off; off = al256(off + bytes); return p; };

  const size_t full = (size_t)BATCH * H0 * W0;          // 1.23M px
  _Float16* img16 = (_Float16*)alloc(full * sizeof(_Float16));
  _Float16* act0  = (_Float16*)alloc(full * 64 * sizeof(_Float16));
  _Float16* act1  = (_Float16*)alloc(full * 64 * sizeof(_Float16));
  _Float16* cPa   = (_Float16*)alloc((size_t)BATCH * 256 * HC * WC * sizeof(_Float16));
  _Float16* cDa   = (_Float16*)alloc((size_t)BATCH * 256 * HC * WC * sizeof(_Float16));
  float* logits   = (float*)alloc((size_t)BATCH * 65  * HC * WC * sizeof(float));
  float* descm    = (float*)alloc((size_t)BATCH * 256 * HC * WC * sizeof(float));
  float* smap     = (float*)alloc(full * sizeof(float));
  float* t0       = (float*)alloc(full * sizeof(float));
  float* t1       = (float*)alloc(full * sizeof(float));
  float* maskb    = (float*)alloc(full * sizeof(float));
  float* suppb    = (float*)alloc(full * sizeof(float));
  float* ssb      = (float*)alloc(full * sizeof(float));
  float* nmsb     = (float*)alloc(full * sizeof(float));
  float* cval     = (float*)alloc((size_t)BATCH * CAND_CAP * sizeof(float));
  int*   cidx     = (int*)  alloc((size_t)BATCH * CAND_CAP * sizeof(int));
  int*   cnt      = (int*)  alloc(16 * sizeof(int));
  _Float16* w16   = (_Float16*)alloc((size_t)1024 * 1024 * sizeof(_Float16)); // per-layer scratch

  // ---- output regions ------------------------------------------------------
  float* o = (float*)d_out;
  float* out_kp     = o;                                    // [4][1024][2]
  float* out_scores = out_kp + (size_t)BATCH * MAX_KP * 2;  // [4][1024]
  float* out_desc   = out_scores + (size_t)BATCH * MAX_KP;  // [4][256][1024]
  float* out_valid  = out_desc + (size_t)BATCH * 256 * MAX_KP; // [4][1024]

  // ---- conv layer launcher -------------------------------------------------
  auto run_conv = [&](const _Float16* in, int layer, int H, int W,
                      _Float16* o16, float* of, int relu) {
    int Cin = CI[layer], Cout = CO[layer], ks = KS[layer];
    int pad = (ks == 3) ? 1 : 0;
    int K = Cin * ks * ks;
    int Kpad = (K + TKC - 1) / TKC * TKC;
    int Mpad = (Cout + TM - 1) / TM * TM;
    int tot = Mpad * Kpad;
    k_convert_weights<<<(tot + 255) / 256, 256, 0, stream>>>(W_[layer], w16, Cout, K, Kpad, Mpad);
    int tilesX = (W + TN - 1) / TN;
    dim3 grid((unsigned)(tilesX * H), (unsigned)(Mpad / TM), BATCH);
    k_conv_wmma<<<grid, 256, 0, stream>>>(in, w16, Bb[layer], o16, of,
                                          Cin, H, W, Cout, ks, pad, relu,
                                          of != nullptr, tilesX);
  };

  // ---- backbone ------------------------------------------------------------
  {
    int n = (int)full;
    k_f32_to_f16<<<(n + 255) / 256, 256, 0, stream>>>(image, img16, n);
  }
  run_conv(img16, 0, H0, W0, act0, nullptr, 1);                 // 1a
  run_conv(act0, 1, H0, W0, act1, nullptr, 1);                  // 1b
  {
    int BC = BATCH * 64, n = BC * (H0 / 2) * (W0 / 2);
    k_pool2<<<(n + 255) / 256, 256, 0, stream>>>(act1, act0, BC, H0, W0);
  }
  run_conv(act0, 2, H0 / 2, W0 / 2, act1, nullptr, 1);          // 2a
  run_conv(act1, 3, H0 / 2, W0 / 2, act0, nullptr, 1);          // 2b
  {
    int BC = BATCH * 64, n = BC * (H0 / 4) * (W0 / 4);
    k_pool2<<<(n + 255) / 256, 256, 0, stream>>>(act0, act1, BC, H0 / 2, W0 / 2);
  }
  run_conv(act1, 4, H0 / 4, W0 / 4, act0, nullptr, 1);          // 3a
  run_conv(act0, 5, H0 / 4, W0 / 4, act1, nullptr, 1);          // 3b
  {
    int BC = BATCH * 128, n = BC * HC * WC;
    k_pool2<<<(n + 255) / 256, 256, 0, stream>>>(act1, act0, BC, H0 / 4, W0 / 4);
  }
  run_conv(act0, 6, HC, WC, act1, nullptr, 1);                  // 4a
  run_conv(act1, 7, HC, WC, act0, nullptr, 1);                  // 4b -> x

  // ---- heads ---------------------------------------------------------------
  run_conv(act0, 8, HC, WC, cPa, nullptr, 1);                   // Pa
  run_conv(cPa, 9, HC, WC, nullptr, logits, 0);                 // Pb -> logits f32
  run_conv(act0, 10, HC, WC, cDa, nullptr, 1);                  // Da
  run_conv(cDa, 11, HC, WC, nullptr, descm, 0);                 // Db -> desc f32

  // ---- softmax + pixel shuffle --------------------------------------------
  {
    int n = BATCH * HC * WC;
    k_softmax_shuffle<<<(n + 255) / 256, 256, 0, stream>>>(logits, smap, HC, WC);
  }

  // ---- simple_nms ----------------------------------------------------------
  int n = (int)full;
  int gb = (n + 255) / 256;
  auto mp9 = [&](const float* src, float* dst) {
    k_mp_row<<<gb, 256, 0, stream>>>(src, t0, BATCH * H0, W0, NMS_R);
    k_mp_col<<<gb, 256, 0, stream>>>(t0, dst, BATCH, H0, W0, NMS_R);
  };
  mp9(smap, t1);
  k_eqmask<<<gb, 256, 0, stream>>>(smap, t1, maskb, n);
  for (int it = 0; it < 2; ++it) {
    mp9(maskb, t1);
    k_supp<<<gb, 256, 0, stream>>>(t1, smap, suppb, ssb, n);
    mp9(ssb, t1);
    k_update<<<gb, 256, 0, stream>>>(ssb, t1, suppb, maskb, n);
  }
  k_final<<<gb, 256, 0, stream>>>(smap, maskb, nmsb, n);

  // ---- extraction ----------------------------------------------------------
  k_zero_ints<<<1, 16, 0, stream>>>(cnt, 16);
  k_compact<<<gb, 256, 0, stream>>>(nmsb, cval, cidx, cnt, H0, W0);
  k_select_topk<<<BATCH, 256, 0, stream>>>(cval, cidx, cnt,
                                           out_kp, out_scores, out_valid, W0);

  // ---- descriptors ---------------------------------------------------------
  {
    int np = BATCH * HC * WC;
    k_norm_map<<<(np + 255) / 256, 256, 0, stream>>>(descm, 256, HC * WC);
  }
  k_sample_desc<<<BATCH * MAX_KP, 256, 0, stream>>>(descm, out_kp, out_desc);
}
